// MSARow_20194936225868
// MI455X (gfx1250) — matrix operand order repaired and checked by reading, not compiled, not run on hardware
//
#include <hip/hip_runtime.h>

#define N_  128
#define L_  256
#define D_  256
#define ZD_ 128
#define H_  8
#define C_  32
#define HC_ 256

typedef __bf16 v16bf __attribute__((ext_vector_type(16)));
typedef float  v8f   __attribute__((ext_vector_type(8)));
typedef unsigned int v8u __attribute__((ext_vector_type(8)));
typedef unsigned int u32x4 __attribute__((ext_vector_type(4)));
typedef int i32x4 __attribute__((ext_vector_type(4)));
typedef int i32x8 __attribute__((ext_vector_type(8)));

#if defined(__has_builtin)
#if __has_builtin(__builtin_amdgcn_tensor_load_to_lds) && \
    __has_builtin(__builtin_amdgcn_s_wait_tensorcnt)
#define HAVE_TDM 1
#endif
#endif
#ifndef HAVE_TDM
#define HAVE_TDM 0
#endif

__device__ __forceinline__ unsigned short f2bf(float f) {
    unsigned int u = __float_as_uint(f);
    unsigned int r = (u + 0x7FFFu + ((u >> 16) & 1u)) >> 16;
    return (unsigned short)r;
}
__device__ __forceinline__ float bf2f(unsigned short u) {
    return __uint_as_float(((unsigned int)u) << 16);
}
__device__ __forceinline__ v16bf pack_ab(uint4 lo, uint4 hi) {
    v8u t;
    t[0] = lo.x; t[1] = lo.y; t[2] = lo.z; t[3] = lo.w;
    t[4] = hi.x; t[5] = hi.y; t[6] = hi.z; t[7] = hi.w;
    return __builtin_bit_cast(v16bf, t);
}
__device__ __forceinline__ float fast_rcp(float x) {
    return __builtin_amdgcn_rcpf(x);   // raw v_rcp_f32, ~1 ulp
}

// ---------------- weight convert + transpose to bf16 ----------------
// dst[c*rows + r] = bf16(src[r*cols + c])
__global__ __launch_bounds__(256) void k_transpose_bf16(
    const float* __restrict__ src, unsigned short* __restrict__ dst,
    int rows, int cols)
{
    int idx = blockIdx.x * blockDim.x + threadIdx.x;
    if (idx < rows * cols) {
        int r = idx / cols, c = idx % cols;
        dst[(size_t)c * rows + r] = f2bf(src[idx]);
    }
}

// ---------------- LayerNorm of m -> bf16 ----------------
__global__ __launch_bounds__(256) void k_ln(
    const float* __restrict__ m, const float* __restrict__ w,
    const float* __restrict__ b, unsigned short* __restrict__ mn)
{
    const int row = blockIdx.x;
    const int t = threadIdx.x;
    float x = m[(size_t)row * D_ + t];
    float s = x, s2 = x * x;
    #pragma unroll
    for (int off = 1; off < 32; off <<= 1) {
        s  += __shfl_xor(s,  off);
        s2 += __shfl_xor(s2, off);
    }
    __shared__ float r1[8], r2[8];
    if ((t & 31) == 0) { r1[t >> 5] = s; r2[t >> 5] = s2; }
    __syncthreads();
    s = 0.f; s2 = 0.f;
    #pragma unroll
    for (int i = 0; i < 8; ++i) { s += r1[i]; s2 += r2[i]; }
    float mu  = s  * (1.0f / D_);
    float var = s2 * (1.0f / D_) - mu * mu;
    float y = (x - mu) * rsqrtf(var + 1e-5f) * w[t] + b[t];
    mn[(size_t)row * D_ + t] = f2bf(y);
}

// ---------------- pair bias: LN(z) @ wz -> [H,L,L] fp32 (h-major) ----------------
__global__ __launch_bounds__(128) void k_bias(
    const float* __restrict__ z, const float* __restrict__ lw,
    const float* __restrict__ lb, const float* __restrict__ wz,
    float* __restrict__ bias)
{
    const int ij = blockIdx.x;        // 0 .. L*L-1
    const int t = threadIdx.x;        // 0 .. 127
    float x = z[(size_t)ij * ZD_ + t];
    float s = x, s2 = x * x;
    #pragma unroll
    for (int off = 1; off < 32; off <<= 1) {
        s  += __shfl_xor(s,  off);
        s2 += __shfl_xor(s2, off);
    }
    __shared__ float r1[4], r2[4];
    if ((t & 31) == 0) { r1[t >> 5] = s; r2[t >> 5] = s2; }
    __syncthreads();
    s = 0.f; s2 = 0.f;
    #pragma unroll
    for (int i = 0; i < 4; ++i) { s += r1[i]; s2 += r2[i]; }
    float mu  = s  * (1.0f / ZD_);
    float var = s2 * (1.0f / ZD_) - mu * mu;
    float y = (x - mu) * rsqrtf(var + 1e-5f) * lw[t] + lb[t];
    __shared__ float yv[128];
    yv[t] = y;
    __syncthreads();
    if (t < H_) {
        float a = 0.f;
        #pragma unroll 8
        for (int k2 = 0; k2 < ZD_; ++k2) a += yv[k2] * wz[k2 * H_ + t];
        bias[(size_t)t * L_ * L_ + ij] = a;    // h-major for coalesced attn reads
    }
}

__device__ __forceinline__ v16bf ld_a(const unsigned short* rowp, int kt, int cb) {
    return pack_ab(*(const uint4*)(rowp + kt * 32 + cb),
                   *(const uint4*)(rowp + kt * 32 + cb + 16));
}
__device__ __forceinline__ v16bf ld_b(const unsigned short* rowp, int kt, int half16) {
    return pack_ab(*(const uint4*)(rowp + kt * 32 + half16 / 2),
                   *(const uint4*)(rowp + kt * 32 + half16 / 2 + 16));
}

// ---------------- q/k/v/g projections (bf16 WMMA GEMM, 32x32/wave) ----------------
__device__ __forceinline__ void store_proj_tile(
    unsigned short* outp, const float* bg, int which,
    int m0t, int n0t, int half, int l15, v8f acc)
{
    #pragma unroll
    for (int r = 0; r < 8; ++r) {
        int row = m0t + r + half * 8;
        int col = n0t + l15;
        float val = acc[r];
        if (which == 3) val = fast_rcp(1.0f + __expf(-(val + bg[col])));
        outp[(size_t)row * HC_ + col] = f2bf(val);
    }
}

__global__ __launch_bounds__(256) void k_proj(
    const unsigned short* __restrict__ mn,
    const unsigned short* __restrict__ wqt, const unsigned short* __restrict__ wkt,
    const unsigned short* __restrict__ wvt, const unsigned short* __restrict__ wgt,
    const float* __restrict__ bg,
    unsigned short* __restrict__ q, unsigned short* __restrict__ k,
    unsigned short* __restrict__ v, unsigned short* __restrict__ g)
{
    const int lane = threadIdx.x & 31;
    const int wvid = threadIdx.x >> 5;
    const int m0 = blockIdx.x * 32;                 // 1024 blocks
    const int nslot = blockIdx.y * 8 + wvid;        // 0..31 over [q|k|v|g]
    const int which = nslot >> 3;                   // 8 slots of 32 cols / matrix
    const int n0 = (nslot & 7) * 32;
    const unsigned short* wt;
    unsigned short* outp;
    if (which == 0)      { wt = wqt; outp = q; }
    else if (which == 1) { wt = wkt; outp = k; }
    else if (which == 2) { wt = wvt; outp = v; }
    else                 { wt = wgt; outp = g; }

    const int half = lane >> 4, l15 = lane & 15;
    const int cb = half ? 8 : 0;
    const int h16 = half * 16;
    const unsigned short* ap0 = mn + (size_t)(m0 + l15) * D_;
    const unsigned short* ap1 = ap0 + (size_t)16 * D_;
    const unsigned short* bp0 = wt + (size_t)(n0 + l15) * D_;
    const unsigned short* bp1 = bp0 + (size_t)16 * D_;

    v8f a00 = {}, a01 = {}, a10 = {}, a11 = {};
    #pragma unroll
    for (int kt = 0; kt < 8; ++kt) {
        v16bf A0 = ld_a(ap0, kt, cb);
        v16bf A1 = ld_a(ap1, kt, cb);
        v16bf B0 = ld_b(bp0, kt, h16);
        v16bf B1 = ld_b(bp1, kt, h16);
        a00 = __builtin_amdgcn_wmma_f32_16x16x32_bf16(false, A0, false, B0, (short)0, a00, false, false);
        a01 = __builtin_amdgcn_wmma_f32_16x16x32_bf16(false, A0, false, B1, (short)0, a01, false, false);
        a10 = __builtin_amdgcn_wmma_f32_16x16x32_bf16(false, A1, false, B0, (short)0, a10, false, false);
        a11 = __builtin_amdgcn_wmma_f32_16x16x32_bf16(false, A1, false, B1, (short)0, a11, false, false);
    }
    store_proj_tile(outp, bg, which, m0,      n0,      half, l15, a00);
    store_proj_tile(outp, bg, which, m0,      n0 + 16, half, l15, a01);
    store_proj_tile(outp, bg, which, m0 + 16, n0,      half, l15, a10);
    store_proj_tile(outp, bg, which, m0 + 16, n0 + 16, half, l15, a11);
}

// ---------------- fused attention per (n,h) ----------------
#define VT_STRIDE 264
#define PS_STRIDE 264

__global__ __launch_bounds__(256) void k_attn(
    const unsigned short* __restrict__ qb, const unsigned short* __restrict__ kb,
    const unsigned short* __restrict__ vb, const unsigned short* __restrict__ gb,
    const float* __restrict__ bias, unsigned short* __restrict__ ob)
{
    __shared__ unsigned short Ks[256 * 32];          // K row-major [key][c]
    __shared__ unsigned short Vt[32 * VT_STRIDE];    // V transposed [c][key]
    __shared__ unsigned short Ps[8 * 16 * PS_STRIDE];// probs per wave [16][256]

    const int n = blockIdx.x / H_;
    const int h = blockIdx.x % H_;
    const int tid = threadIdx.x;
    const int wvid = tid >> 5;
    const int lane = tid & 31;
    const size_t head_base = ((size_t)n * L_) * HC_ + (size_t)h * C_;
    const float* bias_h = bias + (size_t)h * L_ * L_;   // [L,L] for this head

#if HAVE_TDM
    // Tensor Data Mover: one instruction stages the whole K tile
    // ([256 rows x 32 bf16], row stride HC_) from global into LDS.
    if (wvid == 0) {
        unsigned lds_off = (unsigned)(size_t)(void*)Ks;  // flat low bits = LDS byte addr
        unsigned long long ga = (unsigned long long)(size_t)(const void*)(kb + head_base);
        u32x4 g0;
        g0[0] = 1u;                                     // count=1, no gather/iterate
        g0[1] = lds_off;                                // lds_addr
        g0[2] = (unsigned)(ga & 0xFFFFFFFFu);           // global_addr[31:0]
        g0[3] = (unsigned)((ga >> 32) & 0x01FFFFFFu)    // global_addr[56:32]
              | (2u << 30);                             // type=2 ("image")
        i32x8 g1;
        g1[0] = (int)(1u << 16);                        // data_size=1 (2 bytes/elt)
        g1[1] = (int)(((unsigned)HC_ & 0xFFFFu) << 16); // tensor_dim0[15:0]
        g1[2] = (int)(((unsigned)L_  & 0xFFFFu) << 16); // dim0 hi=0 | tensor_dim1[15:0]
        g1[3] = (int)(((unsigned)C_  & 0xFFFFu) << 16); // dim1 hi=0 | tile_dim0=32
        g1[4] = (int)((unsigned)L_ & 0xFFFFu);          // tile_dim1=256 | tile_dim2=0
        g1[5] = (int)(unsigned)HC_;                     // tensor_dim0_stride lo
        g1[6] = 0;                                      // stride hi | dim1_stride lo
        g1[7] = 0;
        i32x4 gz = {};                                  // groups 2/3 unused (2D)
#if __clang_major__ >= 23
        i32x8 gz8 = {};
        __builtin_amdgcn_tensor_load_to_lds(g0, g1, gz, gz, gz8, 0);
#else
        __builtin_amdgcn_tensor_load_to_lds(g0, g1, gz, gz, 0);
#endif
    }
#endif

    // cooperative stage of V (transposed) — and K if no TDM path
    for (int i = tid; i < 256 * 4; i += 256) {
        int row = i >> 2, ch = i & 3;
        size_t gsrc = head_base + (size_t)row * HC_ + ch * 8;
#if !HAVE_TDM
        uint4 dk = *(const uint4*)(kb + gsrc);
        *(uint4*)(Ks + row * 32 + ch * 8) = dk;
#endif
        uint4 dv = *(const uint4*)(vb + gsrc);
        unsigned short e[8];
        *(uint4*)e = dv;
        #pragma unroll
        for (int j = 0; j < 8; ++j)
            Vt[(ch * 8 + j) * VT_STRIDE + row] = e[j];
    }
#if HAVE_TDM
    if (wvid == 0) __builtin_amdgcn_s_wait_tensorcnt(0);
#endif
    __syncthreads();

    const float scale = 0.17677669529663687f;  // 1/sqrt(C)
    const int half = lane >> 4;
    const int l15 = lane & 15;
    const int cb = half ? 8 : 0;

    for (int qt = 0; qt < 2; ++qt) {
        const int q0 = wvid * 32 + qt * 16;
        __builtin_prefetch(gb + head_base + (size_t)q0 * HC_, 0, 1);
        // Q A-operand straight from global (rows contiguous per head)
        const unsigned short* qp = qb + head_base + (size_t)(q0 + l15) * HC_;
        v16bf aq = pack_ab(*(const uint4*)(qp + half * 16), *(const uint4*)(qp + half * 16 + 8));

        // scores: 16 tiles of 16x16 (keys 0..255); bias reads are coalesced
        v8f s[16];
        #pragma unroll
        for (int j = 0; j < 16; ++j) {
            const unsigned short* kp = Ks + (j * 16 + l15) * 32 + half * 16;
            v16bf bk = pack_ab(*(const uint4*)kp, *(const uint4*)(kp + 8));
            v8f zt = {};
            zt = __builtin_amdgcn_wmma_f32_16x16x32_bf16(
                false, aq, false, bk, (short)0, zt, false, false);
            const float* bp = bias_h + (size_t)(q0 + half * 8) * L_ + j * 16 + l15;
            #pragma unroll
            for (int r = 0; r < 8; ++r)
                zt[r] = zt[r] * scale + bp[(size_t)r * L_];
            s[j] = zt;
        }

        // softmax over key dim (row = r + 8*half; cols live in 16-lane halves)
        float rinv[8];
        #pragma unroll
        for (int r = 0; r < 8; ++r) {
            float mx = s[0][r];
            #pragma unroll
            for (int j = 1; j < 16; ++j) mx = fmaxf(mx, s[j][r]);
            mx = fmaxf(mx, __shfl_xor(mx, 1));
            mx = fmaxf(mx, __shfl_xor(mx, 2));
            mx = fmaxf(mx, __shfl_xor(mx, 4));
            mx = fmaxf(mx, __shfl_xor(mx, 8));
            float sum = 0.f;
            #pragma unroll
            for (int j = 0; j < 16; ++j) {
                float p = __expf(s[j][r] - mx);
                s[j][r] = p;
                sum += p;
            }
            sum += __shfl_xor(sum, 1);
            sum += __shfl_xor(sum, 2);
            sum += __shfl_xor(sum, 4);
            sum += __shfl_xor(sum, 8);
            rinv[r] = fast_rcp(sum);
        }

        // stage unnormalized probs (bf16) for P@V A-operand
        unsigned short* pw = Ps + wvid * 16 * PS_STRIDE;
        #pragma unroll
        for (int j = 0; j < 16; ++j) {
            #pragma unroll
            for (int r = 0; r < 8; ++r)
                pw[(r + half * 8) * PS_STRIDE + j * 16 + l15] = f2bf(s[j][r]);
        }

        // O = P @ V  (two 16-wide channel tiles), then 1/sum and gate
        #pragma unroll
        for (int nc = 0; nc < 2; ++nc) {
            v8f acc = {};
            #pragma unroll
            for (int kk = 0; kk < 8; ++kk) {
                const unsigned short* pp = pw + l15 * PS_STRIDE + kk * 32 + cb;
                v16bf apm = pack_ab(*(const uint4*)pp, *(const uint4*)(pp + 16));
                const unsigned short* vp =
                    Vt + (nc * 16 + l15) * VT_STRIDE + kk * 32 + half * 8;
                v16bf bv = pack_ab(*(const uint4*)vp, *(const uint4*)(vp + 16));
                acc = __builtin_amdgcn_wmma_f32_16x16x32_bf16(
                    false, apm, false, bv, (short)0, acc, false, false);
            }
            #pragma unroll
            for (int r = 0; r < 8; ++r) {
                int row = q0 + r + half * 8;
                int c = nc * 16 + l15;
                size_t gi = head_base + (size_t)row * HC_ + c;
                float gval = bf2f(gb[gi]);
                float val = acc[r] * rinv[r] * gval;
                ob[gi] = f2bf(val);
            }
        }
    }
}

// ---------------- output projection + bo (fp32 out, 32x32/wave) ----------------
__global__ __launch_bounds__(256) void k_out(
    const unsigned short* __restrict__ o, const unsigned short* __restrict__ wot,
    const float* __restrict__ bo, float* __restrict__ out)
{
    const int lane = threadIdx.x & 31;
    const int wvid = threadIdx.x >> 5;
    const int m0 = blockIdx.x * 32;     // 1024 blocks
    const int n0 = wvid * 32;           // 8 waves cover D_=256
    const int half = lane >> 4, l15 = lane & 15;
    const int cb = half ? 8 : 0;
    const int h16 = half * 16;
    const unsigned short* ap0 = o + (size_t)(m0 + l15) * HC_;
    const unsigned short* ap1 = ap0 + (size_t)16 * HC_;
    const unsigned short* bp0 = wot + (size_t)(n0 + l15) * HC_;
    const unsigned short* bp1 = bp0 + (size_t)16 * HC_;

    v8f a00 = {}, a01 = {}, a10 = {}, a11 = {};
    #pragma unroll
    for (int kt = 0; kt < 8; ++kt) {
        v16bf A0 = ld_a(ap0, kt, cb);
        v16bf A1 = ld_a(ap1, kt, cb);
        v16bf B0 = ld_b(bp0, kt, h16);
        v16bf B1 = ld_b(bp1, kt, h16);
        a00 = __builtin_amdgcn_wmma_f32_16x16x32_bf16(false, A0, false, B0, (short)0, a00, false, false);
        a01 = __builtin_amdgcn_wmma_f32_16x16x32_bf16(false, A0, false, B1, (short)0, a01, false, false);
        a10 = __builtin_amdgcn_wmma_f32_16x16x32_bf16(false, A1, false, B0, (short)0, a10, false, false);
        a11 = __builtin_amdgcn_wmma_f32_16x16x32_bf16(false, A1, false, B1, (short)0, a11, false, false);
    }
    #pragma unroll
    for (int r = 0; r < 8; ++r) {
        int row0 = m0 + r + half * 8;
        int row1 = row0 + 16;
        int col0 = n0 + l15;
        int col1 = col0 + 16;
        out[(size_t)row0 * D_ + col0] = a00[r] + bo[col0];
        out[(size_t)row0 * D_ + col1] = a01[r] + bo[col1];
        out[(size_t)row1 * D_ + col0] = a10[r] + bo[col0];
        out[(size_t)row1 * D_ + col1] = a11[r] + bo[col1];
    }
}

extern "C" void kernel_launch(void* const* d_in, const int* in_sizes, int n_in,
                              void* d_out, int out_size, void* d_ws, size_t ws_size,
                              hipStream_t stream) {
    (void)in_sizes; (void)n_in; (void)out_size; (void)ws_size;
    const float* m    = (const float*)d_in[0];
    const float* z    = (const float*)d_in[1];
    const float* ln1w = (const float*)d_in[2];
    const float* ln1b = (const float*)d_in[3];
    const float* wq   = (const float*)d_in[4];
    const float* wk   = (const float*)d_in[5];
    const float* wv   = (const float*)d_in[6];
    const float* lnzw = (const float*)d_in[7];
    const float* lnzb = (const float*)d_in[8];
    const float* wz   = (const float*)d_in[9];
    const float* wg   = (const float*)d_in[10];
    const float* bg   = (const float*)d_in[11];
    const float* wo   = (const float*)d_in[12];
    const float* bo   = (const float*)d_in[13];
    float* out = (float*)d_out;

    const size_t NLD = (size_t)N_ * L_ * HC_;       // 8,388,608
    const size_t WSZ = (size_t)D_ * HC_;            // 65,536
    unsigned short* ws16 = (unsigned short*)d_ws;
    unsigned short* mn  = ws16;
    unsigned short* qB  = mn  + NLD;
    unsigned short* kB  = qB  + NLD;
    unsigned short* vB  = kB  + NLD;
    unsigned short* gB  = vB  + NLD;
    unsigned short* oB  = gB  + NLD;
    unsigned short* wqt = oB  + NLD;
    unsigned short* wkt = wqt + WSZ;
    unsigned short* wvt = wkt + WSZ;
    unsigned short* wgt = wvt + WSZ;
    unsigned short* wot = wgt + WSZ;
    float* biasbuf = (float*)(wot + WSZ);

    // weights -> transposed bf16
    k_transpose_bf16<<<256, 256, 0, stream>>>(wq, wqt, D_, HC_);
    k_transpose_bf16<<<256, 256, 0, stream>>>(wk, wkt, D_, HC_);
    k_transpose_bf16<<<256, 256, 0, stream>>>(wv, wvt, D_, HC_);
    k_transpose_bf16<<<256, 256, 0, stream>>>(wg, wgt, D_, HC_);
    k_transpose_bf16<<<256, 256, 0, stream>>>(wo, wot, HC_, D_);

    // LayerNorm m -> bf16
    k_ln<<<N_ * L_, 256, 0, stream>>>(m, ln1w, ln1b, mn);

    // pair bias (h-major)
    k_bias<<<L_ * L_, 128, 0, stream>>>(z, lnzw, lnzb, wz, biasbuf);

    // projections (q,k,v,g): 32x32 per wave, grid covers 4*HC_ columns
    k_proj<<<dim3((N_ * L_) / 32, 4), 256, 0, stream>>>(
        mn, wqt, wkt, wvt, wgt, bg, qB, kB, vB, gB);

    // fused attention per (n,h)
    k_attn<<<N_ * H_, 256, 0, stream>>>(qB, kB, vB, gB, biasbuf, oB);

    // output projection: 32x32 per wave
    k_out<<<dim3((N_ * L_) / 32, 1), 256, 0, stream>>>(oB, wot, bo, out);
}
